// EncoderDecoder_16733192585646
// MI455X (gfx1250) — matrix operand order repaired and checked by reading, not compiled, not run on hardware
//
#include <hip/hip_runtime.h>
#include <hip/hip_bf16.h>

typedef __attribute__((ext_vector_type(16))) __bf16 v16bf;
typedef __attribute__((ext_vector_type(8)))  __bf16 v8bf;
typedef __attribute__((ext_vector_type(8)))  float  v8f;

// ---------- helpers ----------
__device__ __forceinline__ __bf16 f2bf(float f) {
  unsigned u = __float_as_uint(f);
  u += 0x7FFFu + ((u >> 16) & 1u);           // round-to-nearest-even
  unsigned short s = (unsigned short)(u >> 16);
  return __builtin_bit_cast(__bf16, s);
}
__device__ __forceinline__ float sigf(float x) { return 1.0f / (1.0f + __expf(-x)); }

// A-fragment (16x32 bf16): lane r=l&15 is row M, kh=l>>4; elems 0-7 -> K=kh*8+e,
// elems 8-15 -> K=16+kh*8+(e-8).  p points at row + k0 + kh*8.
__device__ __forceinline__ v16bf load_fragA(const __bf16* p) {
  v8bf lo = *reinterpret_cast<const v8bf*>(p);
  v8bf hi = *reinterpret_cast<const v8bf*>(p + 16);
  v16bf f;
#pragma unroll
  for (int i = 0; i < 8; ++i) { f[i] = lo[i]; f[i + 8] = hi[i]; }
  return f;
}
// B-fragment (32x16 bf16): lane n=l&15 is column N; elems 0-7 -> K=kh*8+e,
// elems 8-15 -> K=16+kh*8+(e-8).  p points at W row n + k0 + kh*8  (W stored (N,K) row-major).
__device__ __forceinline__ v16bf load_fragB(const __bf16* p) {
  v8bf lo = *reinterpret_cast<const v8bf*>(p);
  v8bf hi = *reinterpret_cast<const v8bf*>(p + 16);
  v16bf f;
#pragma unroll
  for (int i = 0; i < 8; ++i) { f[i] = lo[i]; f[i + 8] = hi[i]; }
  return f;
}

// ---------- register-blocked wave-tile GEMM ----------
// C[m,n] = sum_k A1[m,k]W1[n,k] (+ A2[m,k]W2[n,k]) + b1[n] + b2[n]
// Each wave computes an MT x NT grid of 16x16 tiles. M % (16*MT) == 0, N % (16*NT) == 0,
// K % 32 == 0 required.
template <int MT, int NT, bool OUT_BF16>
__global__ void __launch_bounds__(256)
wmma_gemm(const __bf16* __restrict__ A1, int K1, const __bf16* __restrict__ W1,
          const __bf16* __restrict__ A2, int K2, const __bf16* __restrict__ W2,
          const float* __restrict__ bias1, const float* __restrict__ bias2,
          void* __restrict__ Cout, int ldc, int M, int N, int tiles) {
  int wave = (int)((blockIdx.x * blockDim.x + threadIdx.x) >> 5);
  if (wave >= tiles) return;                 // whole-wave exit: EXEC all-ones for WMMA
  int lane = (int)(threadIdx.x & 31);
  int nbl = N / (16 * NT);
  int bm = wave / nbl, bn = wave - bm * nbl;
  int r  = lane & 15, kh = lane >> 4;

  v8f acc[MT][NT];
#pragma unroll
  for (int mt = 0; mt < MT; ++mt)
#pragma unroll
    for (int nt = 0; nt < NT; ++nt)
      acc[mt][nt] = (v8f){0.f, 0.f, 0.f, 0.f, 0.f, 0.f, 0.f, 0.f};

  {
    const __bf16* pa[MT];
    const __bf16* pb[NT];
#pragma unroll
    for (int mt = 0; mt < MT; ++mt)
      pa[mt] = A1 + (size_t)(bm * 16 * MT + mt * 16 + r) * K1 + kh * 8;
#pragma unroll
    for (int nt = 0; nt < NT; ++nt)
      pb[nt] = W1 + (size_t)(bn * 16 * NT + nt * 16 + r) * K1 + kh * 8;
    for (int k0 = 0; k0 < K1; k0 += 32) {
      v16bf aF[MT], bF[NT];
#pragma unroll
      for (int mt = 0; mt < MT; ++mt) aF[mt] = load_fragA(pa[mt] + k0);
#pragma unroll
      for (int nt = 0; nt < NT; ++nt) bF[nt] = load_fragB(pb[nt] + k0);
#pragma unroll
      for (int mt = 0; mt < MT; ++mt)
#pragma unroll
        for (int nt = 0; nt < NT; ++nt)
          acc[mt][nt] = __builtin_amdgcn_wmma_f32_16x16x32_bf16(
              false, aF[mt], false, bF[nt], (short)0, acc[mt][nt], false, false);
    }
  }
  if (K2 > 0) {
    const __bf16* pa[MT];
    const __bf16* pb[NT];
#pragma unroll
    for (int mt = 0; mt < MT; ++mt)
      pa[mt] = A2 + (size_t)(bm * 16 * MT + mt * 16 + r) * K2 + kh * 8;
#pragma unroll
    for (int nt = 0; nt < NT; ++nt)
      pb[nt] = W2 + (size_t)(bn * 16 * NT + nt * 16 + r) * K2 + kh * 8;
    for (int k0 = 0; k0 < K2; k0 += 32) {
      v16bf aF[MT], bF[NT];
#pragma unroll
      for (int mt = 0; mt < MT; ++mt) aF[mt] = load_fragA(pa[mt] + k0);
#pragma unroll
      for (int nt = 0; nt < NT; ++nt) bF[nt] = load_fragB(pb[nt] + k0);
#pragma unroll
      for (int mt = 0; mt < MT; ++mt)
#pragma unroll
        for (int nt = 0; nt < NT; ++nt)
          acc[mt][nt] = __builtin_amdgcn_wmma_f32_16x16x32_bf16(
              false, aF[mt], false, bF[nt], (short)0, acc[mt][nt], false, false);
    }
  }

#pragma unroll
  for (int nt = 0; nt < NT; ++nt) {
    int n = bn * 16 * NT + nt * 16 + r;
    float bv = 0.f;
    if (bias1) bv += bias1[n];
    if (bias2) bv += bias2[n];
#pragma unroll
    for (int mt = 0; mt < MT; ++mt) {
#pragma unroll
      for (int e = 0; e < 8; ++e) {          // C layout: m = base + 8*kh + vgpr, n = lane&15
        int m = bm * 16 * MT + mt * 16 + kh * 8 + e;
        float v = acc[mt][nt][e] + bv;
        if (OUT_BF16) ((__bf16*)Cout)[(size_t)m * ldc + n] = f2bf(v);
        else          ((float*)Cout)[(size_t)m * ldc + n] = v;
      }
    }
  }
}

// ---------- f32 -> bf16 convert ----------
__global__ void f32_to_bf16_kernel(const float* __restrict__ src, __bf16* __restrict__ dst, int n) {
  int i = blockIdx.x * blockDim.x + threadIdx.x;
  if (i < n) dst[i] = f2bf(src[i]);
}

// ---------- LSTM pointwise cell (torch gate order i,f,g,o) ----------
__global__ void lstm_cell_kernel(const float* __restrict__ gates,  // (256, 4H)
                                 float* __restrict__ cst,          // (256, H)
                                 __bf16* __restrict__ h_bf,        // (256, H) recurrent input
                                 __bf16* __restrict__ out_slot,    // optional sequence slot
                                 int out_stride, int H) {
  int idx = blockIdx.x * blockDim.x + threadIdx.x;
  if (idx >= 256 * H) return;
  int b = idx / H, u = idx - b * H;
  const float* g = gates + (size_t)b * 4 * H;
  float i  = sigf(g[u]);
  float f  = sigf(g[H + u]);
  float gg = tanhf(g[2 * H + u]);
  float o  = sigf(g[3 * H + u]);
  float cn = f * cst[idx] + i * gg;
  cst[idx] = cn;
  float h = o * tanhf(cn);
  __bf16 hb = f2bf(h);
  h_bf[idx] = hb;
  if (out_slot) out_slot[(size_t)b * out_stride + u] = hb;
}

// ---------- embedding gather (teacher forcing: [SOS, target[:-1]]) ----------
__global__ void embed_fill_kernel(const int* __restrict__ target, const float* __restrict__ emb,
                                  __bf16* __restrict__ in2) {
  int idx = blockIdx.x * blockDim.x + threadIdx.x;   // 20*256*512
  if (idx >= 20 * 256 * 512) return;
  int e = idx & 511;
  int b = (idx >> 9) & 255;
  int t = idx >> 17;
  int tok = (t == 0) ? 1 : target[(t - 1) * 256 + b];
  float v = emb[(size_t)tok * 512 + e];
  in2[(((size_t)(28 + t) * 256 + b) * 1024) + 512 + e] = f2bf(v);
}

// ---------- row softmax over V (one block per row, in place) ----------
__global__ void __launch_bounds__(256) softmax_rows_kernel(float* __restrict__ data, int V) {
  __shared__ float red[256];
  float* p = data + (size_t)blockIdx.x * V;
  int tid = threadIdx.x;
  float mx = -3.4e38f;
  for (int i = tid; i < V; i += 256) mx = fmaxf(mx, p[i]);
  red[tid] = mx; __syncthreads();
  for (int s = 128; s > 0; s >>= 1) { if (tid < s) red[tid] = fmaxf(red[tid], red[tid + s]); __syncthreads(); }
  mx = red[0]; __syncthreads();
  float sum = 0.f;
  for (int i = tid; i < V; i += 256) { float e = __expf(p[i] - mx); p[i] = e; sum += e; }
  red[tid] = sum; __syncthreads();
  for (int s = 128; s > 0; s >>= 1) { if (tid < s) red[tid] += red[tid + s]; __syncthreads(); }
  float inv = 1.0f / red[0];
  for (int i = tid; i < V; i += 256) p[i] *= inv;
}

// ---------- host ----------
extern "C" void kernel_launch(void* const* d_in, const int* in_sizes, int n_in,
                              void* d_out, int out_size, void* d_ws, size_t ws_size,
                              hipStream_t stream) {
  (void)in_sizes; (void)n_in; (void)out_size; (void)ws_size;
  const float* feats  = (const float*)d_in[0];
  const int*   target = (const int*)d_in[1];
  const float* enc_W  = (const float*)d_in[2];
  const float* enc_b  = (const float*)d_in[3];
  const float* emb    = (const float*)d_in[4];
  const float* Wih1   = (const float*)d_in[5];
  const float* Whh1   = (const float*)d_in[6];
  const float* bih1   = (const float*)d_in[7];
  const float* bhh1   = (const float*)d_in[8];
  const float* Wih2   = (const float*)d_in[9];
  const float* Whh2   = (const float*)d_in[10];
  const float* bih2   = (const float*)d_in[11];
  const float* bhh2   = (const float*)d_in[12];
  const float* out_W  = (const float*)d_in[13];
  const float* out_b  = (const float*)d_in[14];

  char* ws = (char*)d_ws;
  size_t off = 0;
  auto alloc = [&](size_t bytes) -> char* {
    char* p = ws + off;
    off += (bytes + 255) & ~(size_t)255;
    return p;
  };
  __bf16* feats_bf = (__bf16*)alloc((size_t)28 * 256 * 2048 * 2);
  __bf16* seq1     = (__bf16*)alloc((size_t)48 * 256 * 512 * 2);   // LSTM1 input (enc out + zero pad)
  __bf16* in2      = (__bf16*)alloc((size_t)48 * 256 * 1024 * 2);  // [h1 | embedding]
  __bf16* out2     = (__bf16*)alloc((size_t)20 * 256 * 512 * 2);   // decoder-range LSTM2 output
  float*  gates    = (float*) alloc((size_t)256 * 2048 * 4);
  float*  cst      = (float*) alloc((size_t)256 * 512 * 4);
  __bf16* hbf      = (__bf16*)alloc((size_t)256 * 512 * 2);
  __bf16* wenc_bf  = (__bf16*)alloc((size_t)512 * 2048 * 2);
  __bf16* wih1_bf  = (__bf16*)alloc((size_t)2048 * 512 * 2);
  __bf16* whh1_bf  = (__bf16*)alloc((size_t)2048 * 512 * 2);
  __bf16* wih2_bf  = (__bf16*)alloc((size_t)2048 * 1024 * 2);
  __bf16* whh2_bf  = (__bf16*)alloc((size_t)2048 * 512 * 2);
  __bf16* outW_bf  = (__bf16*)alloc((size_t)20000 * 512 * 2);

  // zero-init: padded sequences + states
  hipMemsetAsync(seq1, 0, (size_t)48 * 256 * 512 * 2, stream);
  hipMemsetAsync(in2,  0, (size_t)48 * 256 * 1024 * 2, stream);
  hipMemsetAsync(cst,  0, (size_t)256 * 512 * 4, stream);
  hipMemsetAsync(hbf,  0, (size_t)256 * 512 * 2, stream);

  auto cvt = [&](const float* s, __bf16* d, int n) {
    f32_to_bf16_kernel<<<(n + 255) / 256, 256, 0, stream>>>(s, d, n);
  };
  cvt(feats, feats_bf, 28 * 256 * 2048);
  cvt(enc_W, wenc_bf, 512 * 2048);
  cvt(Wih1,  wih1_bf, 2048 * 512);
  cvt(Whh1,  whh1_bf, 2048 * 512);
  cvt(Wih2,  wih2_bf, 2048 * 1024);
  cvt(Whh2,  whh2_bf, 2048 * 512);
  cvt(out_W, outW_bf, 20000 * 512);

  // encoder: (7168 x 2048) @ enc_W^T + enc_b -> seq1 rows 0..7167 (bf16), 32x64 per wave
  {
    int M = 28 * 256, N = 512;
    int tiles = (M / 32) * (N / 64);
    wmma_gemm<2, 4, true><<<(tiles + 7) / 8, 256, 0, stream>>>(
        feats_bf, 2048, wenc_bf, nullptr, 0, nullptr, enc_b, nullptr,
        seq1, 512, M, N, tiles);
  }

  // embedding half of LSTM2 input (decoder frames only; encoder frames stay zero)
  embed_fill_kernel<<<(20 * 256 * 512 + 255) / 256, 256, 0, stream>>>(target, emb, in2);

  const int cellBlocks = (256 * 512 + 255) / 256;
  // recurrent step GEMM config: 16x32 per wave -> 1024 waves/step (occupancy for latency hiding)
  const int stepTiles  = (256 / 16) * (2048 / 32);
  const int stepBlocks = (stepTiles + 7) / 8;

  // LSTM1: gates = x@Wih1^T + h@Whh1^T + b; h -> in2[t][:, 0:512]
  for (int t = 0; t < 48; ++t) {
    const __bf16* x = seq1 + (size_t)t * 256 * 512;
    wmma_gemm<1, 2, false><<<stepBlocks, 256, 0, stream>>>(
        x, 512, wih1_bf, hbf, 512, whh1_bf, bih1, bhh1, gates, 2048, 256, 2048, stepTiles);
    lstm_cell_kernel<<<cellBlocks, 256, 0, stream>>>(
        gates, cst, hbf, in2 + (size_t)t * 256 * 1024, 1024, 512);
  }

  // reset states for LSTM2
  hipMemsetAsync(cst, 0, (size_t)256 * 512 * 4, stream);
  hipMemsetAsync(hbf, 0, (size_t)256 * 512 * 2, stream);

  // LSTM2: input K=1024 ([h1 | emb]), recurrent K=512
  for (int t = 0; t < 48; ++t) {
    const __bf16* x = in2 + (size_t)t * 256 * 1024;
    wmma_gemm<1, 2, false><<<stepBlocks, 256, 0, stream>>>(
        x, 1024, wih2_bf, hbf, 512, whh2_bf, bih2, bhh2, gates, 2048, 256, 2048, stepTiles);
    __bf16* slot = (t >= 28) ? (out2 + (size_t)(t - 28) * 256 * 512) : nullptr;
    lstm_cell_kernel<<<cellBlocks, 256, 0, stream>>>(gates, cst, hbf, slot, 512, 512);
  }

  // logits: (5120 x 512) @ out_W^T + out_b -> d_out (f32, fully overwritten), 32x80 per wave
  {
    int M = 20 * 256, N = 20000;
    int tiles = (M / 32) * (N / 80);
    wmma_gemm<2, 5, false><<<(tiles + 7) / 8, 256, 0, stream>>>(
        out2, 512, outW_bf, nullptr, 0, nullptr, out_b, nullptr,
        d_out, 20000, M, N, tiles);
  }

  // softmax over V per (t, b) row, in place in d_out
  softmax_rows_kernel<<<20 * 256, 256, 0, stream>>>((float*)d_out, 20000);
}
